// SetAbstraction_26998164422712
// MI455X (gfx1250) — matrix operand order repaired and checked by reading, not compiled, not run on hardware
//
#include <hip/hip_runtime.h>
#include <hip/hip_bf16.h>

// ---------------------------------------------------------------------------
// PointNet++ SetAbstraction for MI455X (gfx1250, wave32, WMMA).
// Pipeline: qout | topk (kNN) | GEMM1(f16 WMMA, fused gather) | BN1 stats |
//           GEMM2 | BN2 stats | GEMM3 (fused max/min pool + BN3 sums) |
//           BN3 finalize | output.
// Workspace layout (~154 MB total):
//   [0)            topk idx   : 524288 * int        (2 MB)
//   [2,097,152)    Y1 (f16)   : 524288*64*2         (67 MB)
//   [69,206,016)   Y2 (f16)   : 524288*64*2         (67 MB)
//   [136,314,880)  gmax (f32) : 16384*128*4         (8 MB)
//   [144,703,488)  gmin (f32) : 16384*128*4         (8 MB)
//   [153,092,096)  stats      : 1024 floats
// ---------------------------------------------------------------------------

typedef __attribute__((ext_vector_type(16))) _Float16 v16h;
typedef __attribute__((ext_vector_type(8)))  float    v8f;

#define B_      8
#define N_      16384
#define NPOINT_ 2048
#define NSAMPLE_ 32
#define CIN_    64
#define MTOT_   (B_ * NPOINT_ * NSAMPLE_)   // 524288
#define EPS_    1e-5f

__device__ inline v8f wmma_f16(v16h a, v16h b, v8f c) {
  return __builtin_amdgcn_wmma_f32_16x16x32_f16(false, a, false, b, (short)0, c,
                                                false, false);
}

// A fragment (16x32 f16): lane m=lane&15; lanes 0-15 hold
// K = [0..15], lanes 16-31 hold K = [16..31] (contiguous, matches B).
__device__ inline v16h load_a_frag(const _Float16* lds, int stride, int mbase,
                                   int kbase, int lane) {
  int m = mbase + (lane & 15);
  int kh = lane >> 4;
  const _Float16* p = lds + m * stride + kbase + kh * 16;
  union { v16h v; _Float16 h[16]; } u;
#pragma unroll
  for (int e = 0; e < 16; ++e) u.h[e] = p[e];
  return u.v;
}

// B fragment (32x16 f16): N = lane&15, K = 16*(lane>>4) + e (contiguous).
__device__ inline v16h load_b_frag(const _Float16* ldsW, int stride,
                                   int ncolbase, int kbase, int lane) {
  int n = ncolbase + (lane & 15);
  const _Float16* p = ldsW + n * stride + kbase + (lane >> 4) * 16;
  union { v16h v; _Float16 h[16]; } u;
#pragma unroll
  for (int e = 0; e < 16; ++e) u.h[e] = p[e];
  return u.v;
}

// --------------------------- init BN accumulators --------------------------
__global__ void init_stats_kernel(float* stats) {
  if (threadIdx.x < 512) stats[threadIdx.x] = 0.f;
}

// ------------------------------- q output ----------------------------------
__global__ void qout_kernel(const float* __restrict__ xyz,
                            const int* __restrict__ sidx,
                            float* __restrict__ out) {
  int tid = blockIdx.x * 256 + threadIdx.x;
  if (tid >= B_ * 3 * NPOINT_) return;
  int b = tid / (3 * NPOINT_);
  int rem = tid - b * 3 * NPOINT_;
  int c = rem >> 11, p = rem & (NPOINT_ - 1);
  out[tid] = xyz[(size_t)b * 3 * N_ + c * N_ + sidx[p]];
}

// ------------------------------ kNN top-32 ---------------------------------
// One wave per centroid: per-lane sorted top-32 in registers over a strided
// 512-candidate slice, then LDS dump + 32-round shuffle-min merge.
__global__ __launch_bounds__(256) void topk_kernel(const float* __restrict__ xyz,
                                                   const int* __restrict__ sidx,
                                                   int* __restrict__ topk) {
  __shared__ float sd[8][32][32];
  __shared__ int   si[8][32][32];
  int lane = threadIdx.x & 31, w = threadIdx.x >> 5;
  int g = blockIdx.x * 8 + w;                 // 0 .. B*NPOINT-1
  int b = g >> 11, p = g & (NPOINT_ - 1);
  const float* xb = xyz + (size_t)b * 3 * N_;
  int ci = sidx[p];
  float qx = xb[ci], qy = xb[N_ + ci], qz = xb[2 * N_ + ci];

  float bd[32]; int bi[32];
#pragma unroll
  for (int k = 0; k < 32; ++k) { bd[k] = 3.4e38f; bi[k] = 0; }

  for (int j = lane; j < N_; j += 32) {
    float dx = xb[j] - qx, dy = xb[N_ + j] - qy, dz = xb[2 * N_ + j] - qz;
    float d = dx * dx + dy * dy + dz * dz;
    if (d < bd[31]) {
      bd[31] = d; bi[31] = j;
#pragma unroll
      for (int k = 31; k > 0; --k) {
        if (bd[k] < bd[k - 1]) {
          float td = bd[k]; bd[k] = bd[k - 1]; bd[k - 1] = td;
          int ti = bi[k];   bi[k] = bi[k - 1]; bi[k - 1] = ti;
        }
      }
    }
  }
#pragma unroll
  for (int k = 0; k < 32; ++k) { sd[w][lane][k] = bd[k]; si[w][lane][k] = bi[k]; }
  __syncthreads();

  int ptr = 0;
  for (int s = 0; s < NSAMPLE_; ++s) {
    float bv = sd[w][lane][ptr];
    int bl = lane;
#pragma unroll
    for (int off = 16; off > 0; off >>= 1) {
      float ov = __shfl_xor(bv, off, 32);
      int   ol = __shfl_xor(bl, off, 32);
      if (ov < bv || (ov == bv && ol < bl)) { bv = ov; bl = ol; }
    }
    int widx = __shfl(si[w][lane][ptr], bl, 32);
    if (lane == 0) topk[(size_t)g * NSAMPLE_ + s] = widx;
    if (lane == bl) ptr++;
  }
}

// --------------------------- GEMM1: gather + 67->64 ------------------------
// M-tile 32, K padded 67->96 (3 WMMA K-steps). 8 waves: mhalf = w&1,
// ntile = w>>1 (Cout 64 = 4 N-tiles).
__global__ __launch_bounds__(256) void gemm1_kernel(
    const float* __restrict__ xyz, const float* __restrict__ points,
    const int* __restrict__ sidx, const int* __restrict__ topk,
    const float* __restrict__ W1, const float* __restrict__ b1,
    _Float16* __restrict__ Y1) {
  __shared__ _Float16 sA[32 * 96];
  __shared__ _Float16 sW[64 * 96];
  int t = threadIdx.x;
  int Mbase = blockIdx.x * 32;

  for (int e = t; e < 64 * 96; e += 256) {
    int o = e / 96, c = e - o * 96;
    sW[e] = (_Float16)(c < 67 ? W1[o * 67 + c] : 0.f);
  }
  for (int e = t; e < 32 * 96; e += 256) {
    int r = e / 96, c = e - r * 96;
    int m = Mbase + r;
    float v = 0.f;
    if (c < 67) {
      int b = m >> 16;                    // NPOINT*NSAMPLE = 65536
      int rem = m & 65535;
      int p = rem >> 5;
      int j = topk[m];                    // topk flat layout == m
      if (c < 3) {
        const float* xb = xyz + (size_t)b * 3 * N_ + c * N_;
        v = xb[j] - xb[sidx[p]];
      } else {
        v = points[(size_t)b * CIN_ * N_ + (size_t)(c - 3) * N_ + j];
      }
    }
    sA[e] = (_Float16)v;
  }
  __syncthreads();

  int lane = t & 31, w = t >> 5;
  int mhalf = w & 1, ntile = w >> 1;
  v8f acc = {};
#pragma unroll
  for (int ks = 0; ks < 3; ++ks) {
    v16h a = load_a_frag(sA, 96, 16 * mhalf, ks * 32, lane);
    v16h bf = load_b_frag(sW, 96, ntile * 16, ks * 32, lane);
    acc = wmma_f16(a, bf, acc);
  }
  int ncol = ntile * 16 + (lane & 15);
  float bias = b1[ncol];
  size_t rowb = (size_t)Mbase + 16 * mhalf + 8 * (lane >> 4);
#pragma unroll
  for (int r = 0; r < 8; ++r)
    Y1[(rowb + r) * 64 + ncol] = (_Float16)(acc[r] + bias);
}

// ------------------------- BN stats over f16 activations -------------------
__global__ __launch_bounds__(256) void bnstats_kernel(
    const _Float16* __restrict__ Y, float* __restrict__ sum,
    float* __restrict__ ss) {
  __shared__ float s0[256], s1[256];
  int c = threadIdx.x & 63, rs = threadIdx.x >> 6;
  size_t rowbase = (size_t)blockIdx.x * 1024;     // 512 blocks * 1024 rows
  float a = 0.f, q = 0.f;
  for (int r = rs; r < 1024; r += 4) {
    float v = (float)Y[(rowbase + r) * 64 + c];
    a += v; q += v * v;
  }
  s0[threadIdx.x] = a; s1[threadIdx.x] = q;
  __syncthreads();
  if (rs == 0) {
    a = s0[c] + s0[64 + c] + s0[128 + c] + s0[192 + c];
    q = s1[c] + s1[64 + c] + s1[128 + c] + s1[192 + c];
    atomicAdd(&sum[c], a);
    atomicAdd(&ss[c], q);
  }
}

__global__ void bnfinalize_kernel(const float* sum, const float* ss,
                                  const float* g, const float* be,
                                  float* sc, float* sh, int C) {
  int c = threadIdx.x;
  if (c >= C) return;
  const float inv = 1.f / (float)MTOT_;
  float m = sum[c] * inv;
  float v = ss[c] * inv - m * m;
  float r = rsqrtf(v + EPS_);
  float s = g[c] * r;
  sc[c] = s;
  sh[c] = be[c] - m * s;
}

// --------------------------- GEMM2: bn1+relu, 64->64 -----------------------
__global__ __launch_bounds__(256) void gemm2_kernel(
    const _Float16* __restrict__ Y1, const float* __restrict__ W2,
    const float* __restrict__ b2, const float* __restrict__ sc1,
    const float* __restrict__ sh1, _Float16* __restrict__ Y2) {
  __shared__ _Float16 sA[32 * 64];
  __shared__ _Float16 sW[64 * 64];
  int t = threadIdx.x;
  size_t Mbase = (size_t)blockIdx.x * 32;

  for (int e = t; e < 64 * 64; e += 256) sW[e] = (_Float16)W2[e];
  for (int e = t; e < 32 * 64; e += 256) {
    int r = e >> 6, c = e & 63;
    float v = (float)Y1[(Mbase + r) * 64 + c];
    v = v * sc1[c] + sh1[c];
    sA[e] = (_Float16)(v > 0.f ? v : 0.f);
  }
  __syncthreads();

  int lane = t & 31, w = t >> 5;
  int mhalf = w & 1, ntile = w >> 1;
  v8f acc = {};
#pragma unroll
  for (int ks = 0; ks < 2; ++ks) {
    v16h a = load_a_frag(sA, 64, 16 * mhalf, ks * 32, lane);
    v16h bf = load_b_frag(sW, 64, ntile * 16, ks * 32, lane);
    acc = wmma_f16(a, bf, acc);
  }
  int ncol = ntile * 16 + (lane & 15);
  float bias = b2[ncol];
  size_t rowb = Mbase + 16 * mhalf + 8 * (lane >> 4);
#pragma unroll
  for (int r = 0; r < 8; ++r)
    Y2[(rowb + r) * 64 + ncol] = (_Float16)(acc[r] + bias);
}

// ---------------- GEMM3: bn2+relu, 64->128, fused max/min + BN3 sums -------
// M-tile 32 = exactly one (b,p) group -> block-local max/min pooling.
__global__ __launch_bounds__(256) void gemm3_kernel(
    const _Float16* __restrict__ Y2, const float* __restrict__ W3,
    const float* __restrict__ b3, const float* __restrict__ sc2,
    const float* __restrict__ sh2, float* __restrict__ gmax,
    float* __restrict__ gmin, float* __restrict__ sum3,
    float* __restrict__ ss3) {
  __shared__ _Float16 sA[32 * 64];
  __shared__ _Float16 sW[128 * 64];
  __shared__ float rmax[128], rmin[128], rsum[128], rss[128];
  int t = threadIdx.x;
  size_t Mbase = (size_t)blockIdx.x * 32;

  for (int e = t; e < 128 * 64; e += 256) sW[e] = (_Float16)W3[e];
  for (int e = t; e < 32 * 64; e += 256) {
    int r = e >> 6, c = e & 63;
    float v = (float)Y2[(Mbase + r) * 64 + c];
    v = v * sc2[c] + sh2[c];
    sA[e] = (_Float16)(v > 0.f ? v : 0.f);
  }
  __syncthreads();

  int lane = t & 31, w = t >> 5;
  int mhalf = w & 1, ntbase = w >> 1;          // each wave: ntiles {ntbase, ntbase+4}
  v8f acc[2] = {{}, {}};
#pragma unroll
  for (int q = 0; q < 2; ++q) {
    int nt = ntbase + q * 4;
#pragma unroll
    for (int ks = 0; ks < 2; ++ks) {
      v16h a = load_a_frag(sA, 64, 16 * mhalf, ks * 32, lane);
      v16h bf = load_b_frag(sW, 64, nt * 16, ks * 32, lane);
      acc[q] = wmma_f16(a, bf, acc[q]);
    }
  }

  float wmx[2], wmn[2], wsm[2], wqs[2];
#pragma unroll
  for (int q = 0; q < 2; ++q) {
    int nt = ntbase + q * 4;
    int ncol = nt * 16 + (lane & 15);
    float bias = b3[ncol];
    float mx = -3.4e38f, mn = 3.4e38f, s = 0.f, qs = 0.f;
#pragma unroll
    for (int r = 0; r < 8; ++r) {
      float v = acc[q][r] + bias;
      mx = fmaxf(mx, v); mn = fminf(mn, v); s += v; qs += v * v;
    }
    // combine lane L with L^16 (same column, other 8-row half of the tile)
    mx = fmaxf(mx, __shfl_xor(mx, 16, 32));
    mn = fminf(mn, __shfl_xor(mn, 16, 32));
    s += __shfl_xor(s, 16, 32);
    qs += __shfl_xor(qs, 16, 32);
    wmx[q] = mx; wmn[q] = mn; wsm[q] = s; wqs[q] = qs;
    if (mhalf == 0 && lane < 16) {
      rmax[ncol] = mx; rmin[ncol] = mn; rsum[ncol] = s; rss[ncol] = qs;
    }
  }
  __syncthreads();
  if (mhalf == 1 && lane < 16) {
#pragma unroll
    for (int q = 0; q < 2; ++q) {
      int ncol = (ntbase + q * 4) * 16 + lane;
      float mx = fmaxf(wmx[q], rmax[ncol]);
      float mn = fminf(wmn[q], rmin[ncol]);
      float s = wsm[q] + rsum[ncol];
      float qs = wqs[q] + rss[ncol];
      size_t gidx = (size_t)blockIdx.x * 128 + ncol;   // blockIdx.x = b*2048+p
      gmax[gidx] = mx;
      gmin[gidx] = mn;
      atomicAdd(&sum3[ncol], s);
      atomicAdd(&ss3[ncol], qs);
    }
  }
}

// ----------------------------- final output --------------------------------
// BN3 is per-channel affine; new_points = s>=0 ? s*max+sh : s*min+sh.
__global__ void npout_kernel(const float* __restrict__ gmax,
                             const float* __restrict__ gmin,
                             const float* __restrict__ sc3,
                             const float* __restrict__ sh3,
                             float* __restrict__ out) {
  int tid = blockIdx.x * 256 + threadIdx.x;
  if (tid >= B_ * 128 * NPOINT_) return;
  int b = tid >> 18;                       // 128*2048 = 262144
  int rem = tid & 262143;
  int c = rem >> 11, p = rem & (NPOINT_ - 1);
  size_t g = ((size_t)b * NPOINT_ + p) * 128 + c;
  float s = sc3[c];
  out[tid] = (s >= 0.f) ? s * gmax[g] + sh3[c] : s * gmin[g] + sh3[c];
}

// ---------------------------------------------------------------------------
extern "C" void kernel_launch(void* const* d_in, const int* in_sizes, int n_in,
                              void* d_out, int out_size, void* d_ws,
                              size_t ws_size, hipStream_t stream) {
  (void)in_sizes; (void)n_in; (void)out_size; (void)ws_size;
  const float* xyz   = (const float*)d_in[0];
  const float* points= (const float*)d_in[1];
  const int*   sidx  = (const int*)d_in[2];
  const float* W1 = (const float*)d_in[3];
  const float* b1 = (const float*)d_in[4];
  const float* g1 = (const float*)d_in[5];
  const float* be1= (const float*)d_in[6];
  const float* W2 = (const float*)d_in[7];
  const float* b2 = (const float*)d_in[8];
  const float* g2 = (const float*)d_in[9];
  const float* be2= (const float*)d_in[10];
  const float* W3 = (const float*)d_in[11];
  const float* b3 = (const float*)d_in[12];
  const float* g3 = (const float*)d_in[13];
  const float* be3= (const float*)d_in[14];
  float* out = (float*)d_out;

  char* ws = (char*)d_ws;
  int*      topk = (int*)ws;
  _Float16* Y1   = (_Float16*)(ws + 2097152);
  _Float16* Y2   = (_Float16*)(ws + 69206016);
  float*    gmax = (float*)(ws + 136314880);
  float*    gmin = (float*)(ws + 144703488);
  float*    stats= (float*)(ws + 153092096);
  float *sum1 = stats,       *ss1 = stats + 64;
  float *sum2 = stats + 128, *ss2 = stats + 192;
  float *sum3 = stats + 256, *ss3 = stats + 384;
  float *sc1 = stats + 512, *sh1 = stats + 576;
  float *sc2 = stats + 640, *sh2 = stats + 704;
  float *sc3 = stats + 768, *sh3 = stats + 896;

  const int MBLK = MTOT_ / 32;   // 16384 GEMM blocks

  init_stats_kernel<<<1, 512, 0, stream>>>(stats);
  qout_kernel<<<(B_ * 3 * NPOINT_ + 255) / 256, 256, 0, stream>>>(xyz, sidx, out);
  topk_kernel<<<(B_ * NPOINT_) / 8, 256, 0, stream>>>(xyz, sidx, topk);
  gemm1_kernel<<<MBLK, 256, 0, stream>>>(xyz, points, sidx, topk, W1, b1, Y1);
  bnstats_kernel<<<512, 256, 0, stream>>>(Y1, sum1, ss1);
  bnfinalize_kernel<<<1, 64, 0, stream>>>(sum1, ss1, g1, be1, sc1, sh1, 64);
  gemm2_kernel<<<MBLK, 256, 0, stream>>>(Y1, W2, b2, sc1, sh1, Y2);
  bnstats_kernel<<<512, 256, 0, stream>>>(Y2, sum2, ss2);
  bnfinalize_kernel<<<1, 64, 0, stream>>>(sum2, ss2, g2, be2, sc2, sh2, 64);
  gemm3_kernel<<<MBLK, 256, 0, stream>>>(Y2, W3, b3, sc2, sh2, gmax, gmin,
                                         sum3, ss3);
  bnfinalize_kernel<<<1, 128, 0, stream>>>(sum3, ss3, g3, be3, sc3, sh3, 128);
  npout_kernel<<<(B_ * 128 * NPOINT_ + 255) / 256, 256, 0, stream>>>(
      gmax, gmin, sc3, sh3, out + B_ * 3 * NPOINT_);
}
